// VFNet_7670811590760
// MI455X (gfx1250) — matrix operand order repaired and checked by reading, not compiled, not run on hardware
//
#include <hip/hip_runtime.h>
#include <hip/hip_bf16.h>
#include <math.h>

#define B_    2
#define NCAM  6
#define C_IN  64
#define H_    44
#define W_    80
#define VX    100
#define VY    100
#define VZ    20
#define N_VOX (VX*VY*VZ)
#define HW    (H_*W_)
#define C1    ((C_IN+1)*VZ)   /* 1300 */
#define CT1   41              /* ceil(1300/32) */
#define C1PAD (CT1*32)        /* 1312 */
#define CT2   8               /* 256/32 */
#define NOUT  256

typedef __attribute__((ext_vector_type(16))) __bf16 v16bf;
typedef __attribute__((ext_vector_type(8)))  float  v8f;

union FragB { v16bf v; unsigned u[8]; };
union FragC { v8f   v; float    f[8]; };

__device__ __forceinline__ unsigned short f2bf(float f) {
    unsigned u = __float_as_uint(f);
    u += 0x7FFFu + ((u >> 16) & 1u);   // round-to-nearest-even
    return (unsigned short)(u >> 16);
}

// ---------------------------------------------------------------------------
// Stage 1: voxel projection + bilinear sampling + camera aggregation.
// One thread per (b, voxel). Output: bf16 volume (B, C1PAD, 100, 100),
// channel = f*VZ + z; channels [1300, 1312) zero-filled padding.
// ---------------------------------------------------------------------------
__global__ __launch_bounds__(256) void project_sample(
    const float* __restrict__ mask,      // (B,NCAM,1,H,W)
    const float* __restrict__ intr,      // (B,NCAM,4,4)
    const float* __restrict__ invext,    // (B,NCAM,4,4)
    const float* __restrict__ feats,     // (B,NCAM,64,H,W)
    unsigned short* __restrict__ vf)     // (B,C1PAD,100,100) bf16
{
    int idx = blockIdx.x * blockDim.x + threadIdx.x;
    if (idx >= B_ * N_VOX) return;
    int b = idx / N_VOX;
    int n = idx - b * N_VOX;
    int z = n / (VY * VX);
    int r = n - z * VY * VX;
    int y = r / VX;
    int x = r - y * VX;

    float px = -50.0f + (float)x;
    float py = -50.0f + (float)y;
    float pz = -15.0f + 1.5f * (float)z;

    float wts[NCAM][4];
    int   ofs[NCAM][4];
    float zcv[NCAM];
    int   camb[NCAM];
    float cnt = 0.0f;

#pragma unroll
    for (int cam = 0; cam < NCAM; ++cam) {
        const float* E  = invext + (size_t)(b * NCAM + cam) * 16;
        float v0 = E[0]*px + E[1]*py + E[2]*pz + E[3];
        float v1 = E[4]*px + E[5]*py + E[6]*pz + E[7];
        float v2 = E[8]*px + E[9]*py + E[10]*pz + E[11];
        const float* Km = intr + (size_t)(b * NCAM + cam) * 16;
        float c0 = Km[0]*v0 + Km[1]*v1 + Km[2]*v2;
        float c1 = Km[4]*v0 + Km[5]*v1 + Km[6]*v2;
        float c2 = Km[8]*v0 + Km[9]*v1 + Km[10]*v2;
        float zi = 1.0f / (c2 + 1e-8f);
        float p0 = c0 * zi;
        float p1 = c1 * zi;
        float gx = (p0 / (float)(W_ - 1) - 0.5f) * 2.0f;
        float gy = (p1 / (float)(H_ - 1) - 0.5f) * 2.0f;

        // nearest-mask sample (round half-even == rintf default RNE)
        float ixr = rintf((gx + 1.0f) * 0.5f * (float)(W_ - 1));
        float iyr = rintf((gy + 1.0f) * 0.5f * (float)(H_ - 1));
        float mok = (ixr >= 0.f && ixr <= (float)(W_-1) &&
                     iyr >= 0.f && iyr <= (float)(H_-1)) ? 1.0f : 0.0f;
        int mxi = min(max((int)ixr, 0), W_ - 1);
        int myi = min(max((int)iyr, 0), H_ - 1);
        float mval = mask[(size_t)(b * NCAM + cam) * HW + myi * W_ + mxi] * mok;

        bool msel = mval > 0.5f;
        bool mdep = v2 > 0.0f;
        bool moob = !(fabsf(gx) > 1.0f || fabsf(gy) > 1.0f);
        float valid = (msel && mdep && moob) ? 1.0f : 0.0f;
        cnt += valid;
        zcv[cam] = (v2 * 0.01f) * valid;
        camb[cam] = (b * NCAM + cam) * C_IN * HW;

        float fx = (gx + 1.0f) * 0.5f * (float)(W_ - 1);
        float fy = (gy + 1.0f) * 0.5f * (float)(H_ - 1);
        float x0 = floorf(fx), y0 = floorf(fy);
        float x1 = x0 + 1.0f,  y1 = y0 + 1.0f;
        float cx[4] = {x0, x1, x0, x1};
        float cy[4] = {y0, y0, y1, y1};
        float cw[4] = {(x1-fx)*(y1-fy), (fx-x0)*(y1-fy),
                       (x1-fx)*(fy-y0), (fx-x0)*(fy-y0)};
#pragma unroll
        for (int j = 0; j < 4; ++j) {
            float ok = (cx[j] >= 0.f && cx[j] <= (float)(W_-1) &&
                        cy[j] >= 0.f && cy[j] <= (float)(H_-1)) ? 1.0f : 0.0f;
            int xi = min(max((int)cx[j], 0), W_ - 1);
            int yi = min(max((int)cy[j], 0), H_ - 1);
            ofs[cam][j] = yi * W_ + xi;
            wts[cam][j] = cw[j] * ok * valid;
        }
    }

    float inv = 1.0f / (cnt + 1e-7f);
    size_t vbase = (size_t)b * C1PAD * (VY * VX);
    int spat = y * VX + x;

    for (int f = 0; f < C_IN; ++f) {
        float s = 0.0f;
#pragma unroll
        for (int cam = 0; cam < NCAM; ++cam) {
            const float* img = feats + (size_t)camb[cam] + (size_t)f * HW;
            s += wts[cam][0] * img[ofs[cam][0]]
               + wts[cam][1] * img[ofs[cam][1]]
               + wts[cam][2] * img[ofs[cam][2]]
               + wts[cam][3] * img[ofs[cam][3]];
        }
        vf[vbase + (size_t)(f * VZ + z) * (VY * VX) + spat] = f2bf(s * inv);
    }
    float sz = 0.0f;
#pragma unroll
    for (int cam = 0; cam < NCAM; ++cam) sz += zcv[cam];
    vf[vbase + (size_t)(C_IN * VZ + z) * (VY * VX) + spat] = f2bf(sz * inv);
    // zero-fill padding channels [C1, C1PAD): one element per (z<pad) voxel
    if (z < C1PAD - C1)
        vf[vbase + (size_t)(C1 + z) * (VY * VX) + spat] = 0;
}

// ---------------------------------------------------------------------------
// Stage 2: weight swizzle to the exact LDS tile image.
// K is reordered: k = r*Cpad + c  (r = ky*3+kx outer, channel inner).
// Layout: [r*CT + ct][bn(=n/64)][n%64][kk(=c%32)], bf16. Zero-pad c >= C.
// ---------------------------------------------------------------------------
__global__ __launch_bounds__(256) void swizzle_w(
    const float* __restrict__ w,          // (256, C, 3, 3)
    unsigned short* __restrict__ ws,      // 9*CT*4*64*32
    int C, int CT, int total)
{
    int i = blockIdx.x * blockDim.x + threadIdx.x;
    if (i >= total) return;
    int kk  = i & 31;
    int n   = (i >> 5) & 63;
    int bn  = (i >> 11) & 3;
    int blk = i >> 13;              // r*CT + ct
    int r   = blk / CT;
    int ct  = blk - r * CT;
    int c   = ct * 32 + kk;
    int ng  = bn * 64 + n;
    unsigned short v = 0;
    if (c < C) v = f2bf(w[((size_t)ng * C + c) * 9 + r]);
    ws[i] = v;
}

// ---------------------------------------------------------------------------
// Stage 3/4: implicit-GEMM conv (3x3, stride 2, pad 1) with bf16 WMMA.
// Block = 256 threads (8 waves), tile 64(M) x 64(N), K-step 32, double-
// buffered LDS (one barrier per K-step). Outer loop over the 3x3 tap r
// hoists all bounds math; A gathers are branch-free (clamped base + mask).
// X must have Cpad = CT*32 channels (zero padded).
// ---------------------------------------------------------------------------
#define TM 64
#define TN 64
#define TK 32

__global__ __launch_bounds__(256) void conv_gemm_wmma(
    const unsigned short* __restrict__ X,    // (B, Cpad, IH, IW) bf16
    const unsigned short* __restrict__ Wswz, // swizzled weights
    const float* __restrict__ bias,          // (256)
    unsigned short* __restrict__ out_bf,     // bf16 out (or null)
    float* __restrict__ out_f32,             // f32 out (or null)
    int CT, int Cpad, int IH, int IW, int OH, int OW, int M_total)
{
    __shared__ __align__(16) unsigned short sA[2][TM * TK]; // m-major, stride TK
    __shared__ __align__(16) unsigned short sB[2][TN * TK]; // n-major, stride TK

    const int tid  = threadIdx.x;
    const int lane = tid & 31;
    const int wid  = tid >> 5;
    const int hi   = lane >> 4;
    const int m0   = blockIdx.x * TM;
    const int n0   = blockIdx.y * TN;
    const int mw   = (wid & 3) * 16;
    const int nw   = (wid >> 2) * 32;
    const int spat = OH * OW;
    const int IHW  = IH * IW;
    const int kpi  = tid & 15;          // A k-pair index: kk = 2*kpi
    const int row0 = tid >> 4;          // A rows: row0 + 16*i
    const long CSTEP = (long)32 * IHW;  // channel-tile stride in elements

    // hoisted per-thread M-row decomposition (loop invariant)
    int rb[4], roy[4], rox[4], rok[4];
#pragma unroll
    for (int i = 0; i < 4; ++i) {
        int m = m0 + row0 + i * 16;
        rok[i] = (m < M_total);
        int mm = rok[i] ? m : 0;
        int b  = mm / spat;
        int rr = mm - b * spat;
        roy[i] = rr / OW;
        rox[i] = rr - roy[i] * OW;
        rb[i]  = b;
    }

    FragC acc0, acc1;
#pragma unroll
    for (int i = 0; i < 8; ++i) { acc0.f[i] = 0.0f; acc1.f[i] = 0.0f; }

    int buf = 0;
    for (int r = 0; r < 9; ++r) {
        const int ky = r / 3;
        const int kx = r - ky * 3;
        // per-row tap geometry, computed once per r; invalid rows get a
        // clamped (always in-bounds) base address and a zero mask.
        const unsigned short* ap[4];
        unsigned amask[4];
#pragma unroll
        for (int i = 0; i < 4; ++i) {
            int iy = roy[i] * 2 - 1 + ky;
            int ix = rox[i] * 2 - 1 + kx;
            int ok = rok[i] && iy >= 0 && iy < IH && ix >= 0 && ix < IW;
            amask[i] = ok ? 0xFFFFFFFFu : 0u;
            long base = ok ? ((long)rb[i] * Cpad * IHW + (long)iy * IW + ix
                              + (long)(2 * kpi) * IHW)
                           : 0;
            ap[i] = X + base;
        }
        const uint4* bsrc = (const uint4*)(Wswz
            + (((size_t)r * CT) * 4 + blockIdx.y) * (size_t)(TN * TK));

        for (int ct = 0; ct < CT; ++ct) {
            // ---- A tile: batched branch-free gathers, then pack+store ----
            unsigned lo[4], h2[4];
#pragma unroll
            for (int i = 0; i < 4; ++i) {
                lo[i] = ap[i][0];
                h2[i] = ap[i][IHW];
                ap[i] += CSTEP;
            }
            unsigned* sAu = (unsigned*)sA[buf];
#pragma unroll
            for (int i = 0; i < 4; ++i)
                sAu[(row0 + i * 16) * (TK / 2) + kpi] =
                    (lo[i] | (h2[i] << 16)) & amask[i];
            // ---- B tile: straight 4KB copy (b128 in, b128 to LDS) ----
            ((uint4*)sB[buf])[tid] = bsrc[tid];
            if (ct + 1 < CT)
                __builtin_prefetch((const char*)(bsrc + 256) + (tid & 63) * 64, 0, 0);
            __syncthreads();

            // ---- fragments (ISA 16-bit A/B VGPR layouts) ----
            const unsigned* pA = (const unsigned*)sA[buf];
            const unsigned* pB = (const unsigned*)sB[buf];
            const int ml  = mw + (lane & 15);
            const int nl0 = nw + (lane & 15);
            FragB fa, fb0, fb1;
#pragma unroll
            for (int v = 0; v < 8; ++v) {
                int ka = v + ((v >= 4) ? 4 : 0) + hi * 4; // A: k0=2v+(v>=4?8:0)+hi*8
                fa.u[v] = pA[ml * (TK / 2) + ka];
                int kb = v + ((v >= 4) ? 4 : 0) + hi * 4; // B: k0=2v+(v>=4?8:0)+hi*8
                fb0.u[v] = pB[nl0 * (TK / 2) + kb];
                fb1.u[v] = pB[(nl0 + 16) * (TK / 2) + kb];
            }
            acc0.v = __builtin_amdgcn_wmma_f32_16x16x32_bf16(
                false, fa.v, false, fb0.v, (short)0, acc0.v, false, false);
            acc1.v = __builtin_amdgcn_wmma_f32_16x16x32_bf16(
                false, fa.v, false, fb1.v, (short)0, acc1.v, false, false);
            buf ^= 1;
            bsrc += 4 * (TN * TK) / 8;   // next (r,ct) block, same bn
        }
    }

    // ---- epilogue: bias + ELU, scatter to (B, 256, OH, OW) ----
    const int nA = n0 + nw + (lane & 15);
    const int nB = nA + 16;
    const float biasA = bias[nA];
    const float biasB = bias[nB];
#pragma unroll
    for (int v = 0; v < 8; ++v) {
        int m = m0 + mw + v + hi * 8;
        if (m >= M_total) continue;
        int b  = m / spat;
        int rr = m - b * spat;
        float vA = acc0.f[v] + biasA;
        float vB = acc1.f[v] + biasB;
        vA = (vA > 0.0f) ? vA : (__expf(vA) - 1.0f);
        vB = (vB > 0.0f) ? vB : (__expf(vB) - 1.0f);
        size_t oA = ((size_t)b * NOUT + nA) * spat + rr;
        size_t oB = ((size_t)b * NOUT + nB) * spat + rr;
        if (out_bf) {
            out_bf[oA] = f2bf(vA);
            out_bf[oB] = f2bf(vB);
        } else {
            out_f32[oA] = vA;
            out_f32[oB] = vB;
        }
    }
}

// ---------------------------------------------------------------------------
extern "C" void kernel_launch(void* const* d_in, const int* in_sizes, int n_in,
                              void* d_out, int out_size, void* d_ws, size_t ws_size,
                              hipStream_t stream)
{
    const float* mask   = (const float*)d_in[0];
    const float* intr   = (const float*)d_in[1];
    const float* invext = (const float*)d_in[4];
    const float* feats  = (const float*)d_in[5];
    const float* w1     = (const float*)d_in[6];
    const float* b1     = (const float*)d_in[7];
    const float* w2     = (const float*)d_in[8];
    const float* b2     = (const float*)d_in[9];

    const int M1 = B_ * 50 * 50;  // 5000
    const int M2 = B_ * 25 * 25;  // 1250
    const int W1TOT = 9 * CT1 * 4 * 64 * 32;   // 3,022,848
    const int W2TOT = 9 * CT2 * 4 * 64 * 32;   //   589,824

    size_t off = 0;
    auto take = [&](size_t bytes) {
        void* p = (char*)d_ws + off;
        off += (bytes + 255) & ~(size_t)255;
        return p;
    };
    unsigned short* vf   = (unsigned short*)take((size_t)B_ * C1PAD * VY * VX * 2);
    unsigned short* w1s  = (unsigned short*)take((size_t)W1TOT * 2);
    unsigned short* w2s  = (unsigned short*)take((size_t)W2TOT * 2);
    unsigned short* act1 = (unsigned short*)take((size_t)B_ * NOUT * 2500 * 2);
    (void)ws_size; (void)in_sizes; (void)n_in; (void)out_size;

    // Stage 1: projection + sampling -> bf16 voxel volume (padded channels)
    {
        int total = B_ * N_VOX;
        project_sample<<<(total + 255) / 256, 256, 0, stream>>>(
            mask, intr, invext, feats, vf);
    }
    // Stage 2: weight swizzles
    swizzle_w<<<(W1TOT + 255) / 256, 256, 0, stream>>>(w1, w1s, C1, CT1, W1TOT);
    swizzle_w<<<(W2TOT + 255) / 256, 256, 0, stream>>>(w2, w2s, NOUT, CT2, W2TOT);

    // Stage 3: conv1 (1300 -> 256, 100x100 -> 50x50), bf16 out + ELU
    {
        dim3 grid((M1 + TM - 1) / TM, NOUT / TN);
        conv_gemm_wmma<<<grid, 256, 0, stream>>>(
            vf, w1s, b1, act1, nullptr, CT1, C1PAD, 100, 100, 50, 50, M1);
    }
    // Stage 4: conv2 (256 -> 256, 50x50 -> 25x25), f32 out + ELU
    {
        dim3 grid((M2 + TM - 1) / TM, NOUT / TN);
        conv_gemm_wmma<<<grid, 256, 0, stream>>>(
            act1, w2s, b2, nullptr, (float*)d_out, CT2, NOUT, 50, 50, 25, 25, M2);
    }
}
